// GatedGraphConvolution_50835232916339
// MI455X (gfx1250) — matrix operand order repaired and checked by reading, not compiled, not run on hardware
//
#include <hip/hip_runtime.h>
#include <stdint.h>

#define DFEAT 128

typedef __attribute__((ext_vector_type(16))) __bf16 bf16x16;
typedef __attribute__((ext_vector_type(2)))  __bf16 bf16x2;
typedef __attribute__((ext_vector_type(8)))  float  f32x8;
typedef __attribute__((ext_vector_type(2)))  float  f32x2;

union Frag {
  bf16x16 v;
  uint32_t u[8];
};

// fp32 pair -> packed bf16 pair via hardware V_CVT_PK_BF16_F32
__device__ __forceinline__ uint32_t pack2bf(float lo, float hi) {
  f32x2 f = {lo, hi};
  bf16x2 b = __builtin_convertvector(f, bf16x2);
  return __builtin_bit_cast(uint32_t, b);
}

__device__ __forceinline__ f32x8 wmma_bf16(Frag a, Frag b, f32x8 c) {
  return __builtin_amdgcn_wmma_f32_16x16x32_bf16(
      /*neg_a=*/false, a.v, /*neg_b=*/false, b.v,
      /*c_mod=*/(short)0, c, /*reuse_a=*/false, /*reuse_b=*/false);
}

// ---- A-fragment (16x32 bf16) from row-major fp32 [N, 128] ----------------
// ISA layout: lanes 0-15 hold row M=lane, K elems {0..7,16..23} of K-block;
// lanes 16-31 hold row M=lane-16, K elems {8..15,24..31}.
__device__ __forceinline__ Frag load_afrag_f32(const float* base, int r0,
                                               int kb, int n_nodes) {
  int lane = threadIdx.x & 31;
  int row  = r0 + (lane & 15);
  if (row >= n_nodes) row = n_nodes - 1;
  int koff = kb * 32 + ((lane >> 4) << 3);  // 0 or 8
  const float* p = base + (size_t)row * DFEAT + koff;
  float4 a0 = *(const float4*)(p);
  float4 a1 = *(const float4*)(p + 4);
  float4 b0 = *(const float4*)(p + 16);
  float4 b1 = *(const float4*)(p + 20);
  Frag f;
  f.u[0] = pack2bf(a0.x, a0.y); f.u[1] = pack2bf(a0.z, a0.w);
  f.u[2] = pack2bf(a1.x, a1.y); f.u[3] = pack2bf(a1.z, a1.w);
  f.u[4] = pack2bf(b0.x, b0.y); f.u[5] = pack2bf(b0.z, b0.w);
  f.u[6] = pack2bf(b1.x, b1.y); f.u[7] = pack2bf(b1.z, b1.w);
  return f;
}

// Same, but applies z = relu(v*scale[c] + shift[c]) per column (fused BN+ReLU)
__device__ __forceinline__ Frag load_afrag_bnrelu(const float* base, int r0,
                                                  int kb, int n_nodes,
                                                  const float* sa,
                                                  const float* sc) {
  int lane = threadIdx.x & 31;
  int row  = r0 + (lane & 15);
  if (row >= n_nodes) row = n_nodes - 1;
  int koff = kb * 32 + ((lane >> 4) << 3);
  const float* p = base + (size_t)row * DFEAT + koff;
  float v[16];
#pragma unroll
  for (int j = 0; j < 8; ++j) {
    int cA = koff + j;
    int cB = koff + 16 + j;
    v[j]     = fmaxf(p[j]      * sa[cA] + sc[cA], 0.0f);
    v[8 + j] = fmaxf(p[16 + j] * sa[cB] + sc[cB], 0.0f);
  }
  Frag f;
#pragma unroll
  for (int j = 0; j < 4; ++j) {
    f.u[j]     = pack2bf(v[2 * j],     v[2 * j + 1]);
    f.u[4 + j] = pack2bf(v[8 + 2 * j], v[9 + 2 * j]);
  }
  return f;
}

// A-fragment from per-wave LDS tile [16][128] fp32
__device__ __forceinline__ Frag load_afrag_lds(const float* lds, int kb) {
  int lane = threadIdx.x & 31;
  int row  = lane & 15;
  int koff = kb * 32 + ((lane >> 4) << 3);
  const float* p = lds + row * DFEAT + koff;
  float4 a0 = *(const float4*)(p);
  float4 a1 = *(const float4*)(p + 4);
  float4 b0 = *(const float4*)(p + 16);
  float4 b1 = *(const float4*)(p + 20);
  Frag f;
  f.u[0] = pack2bf(a0.x, a0.y); f.u[1] = pack2bf(a0.z, a0.w);
  f.u[2] = pack2bf(a1.x, a1.y); f.u[3] = pack2bf(a1.z, a1.w);
  f.u[4] = pack2bf(b0.x, b0.y); f.u[5] = pack2bf(b0.z, b0.w);
  f.u[6] = pack2bf(b1.x, b1.y); f.u[7] = pack2bf(b1.z, b1.w);
  return f;
}

// ---- B-fragment (32x16 bf16) from pre-packed bf16 weights ---------------
// Weight W is [128 out][128 in] bf16 pairs (uint32[128][64]); B[k,n]=W[n,k].
// ISA layout: lane n<16 = col n0+n holds K {0..7,16..23}; lane n+16 holds K {8..15,24..31}.
__device__ __forceinline__ Frag load_bfrag(const uint32_t* wb, int n0, int kb) {
  int lane = threadIdx.x & 31;
  int n = n0 + (lane & 15);
  int kstart = kb * 32 + ((lane >> 4) << 3);  // 0 or 8
  const uint32_t* p = wb + n * (DFEAT / 2) + (kstart >> 1);
  uint4 q0 = *(const uint4*)(p);
  uint4 q1 = *(const uint4*)(p + 8);
  Frag f;
  f.u[0] = q0.x; f.u[1] = q0.y; f.u[2] = q0.z; f.u[3] = q0.w;
  f.u[4] = q1.x; f.u[5] = q1.y; f.u[6] = q1.z; f.u[7] = q1.w;
  return f;
}

// ======================= K0: fp32 -> packed bf16 weights ==================
__global__ __launch_bounds__(256) void cvt_bf16_kernel(const float* __restrict__ src,
                                                       uint32_t* __restrict__ dst,
                                                       int npairs) {
  int i = blockIdx.x * 256 + threadIdx.x;
  if (i < npairs) {
    float2 f = ((const float2*)src)[i];
    dst[i] = pack2bf(f.x, f.y);
  }
}

// ======================= K1: zero fill ====================================
__global__ __launch_bounds__(256) void zero_kernel(float* __restrict__ p, int n) {
  int i = blockIdx.x * 256 + threadIdx.x;
  if (i < n) p[i] = 0.0f;
}

// ======================= K2: SpMM scatter (edge-parallel) =================
// One wave per edge; lane l handles 4 contiguous features. x rows are
// L2-resident (51 MB < 192 MB L2), atomics land in L2.
__global__ __launch_bounds__(256) void spmm_kernel(const float* __restrict__ x,
                                                   const int* __restrict__ erow,
                                                   const int* __restrict__ ecol,
                                                   const float* __restrict__ eval_,
                                                   float* __restrict__ agg,
                                                   int n_edges) {
  int t = blockIdx.x * 256 + threadIdx.x;
  int e = t >> 5;
  if (e >= n_edges) return;
  int lane = t & 31;
  int r = erow[e];
  int c = ecol[e];
  float v = eval_[e];
  const float4 xv = *(const float4*)(x + (size_t)c * DFEAT + lane * 4);
  float* dst = agg + (size_t)r * DFEAT + lane * 4;
  unsafeAtomicAdd(dst + 0, v * xv.x);
  unsafeAtomicAdd(dst + 1, v * xv.y);
  unsafeAtomicAdd(dst + 2, v * xv.z);
  unsafeAtomicAdd(dst + 3, v * xv.w);
}

// ======================= K3: fused gate (6 GEMMs + blend + BN1 stats) =====
// One wave per 16 rows. Phase 1: rl = relu(agg@W2^T + x@U2^T + b), rlx->LDS.
// Phase 2: ul, fl, out = ul*fl + (1-ul)*x; accumulate BN1 sum/sumsq.
__global__ __launch_bounds__(32) void gate_kernel(
    const float* __restrict__ x, const float* __restrict__ agg,
    const uint32_t* __restrict__ w1, const uint32_t* __restrict__ u1,
    const uint32_t* __restrict__ w2, const uint32_t* __restrict__ u2,
    const uint32_t* __restrict__ w3, const uint32_t* __restrict__ u3,
    const float* __restrict__ W1b, const float* __restrict__ U1b,
    const float* __restrict__ W2b, const float* __restrict__ U2b,
    const float* __restrict__ W3b, const float* __restrict__ U3b,
    float* __restrict__ out, float* __restrict__ bn_sum,
    float* __restrict__ bn_sq, int n_nodes) {
  __shared__ float rlx[16 * DFEAT];  // per-wave 8 KB
  int r0 = blockIdx.x * 16;
  if (r0 >= n_nodes) return;
  int lane = threadIdx.x & 31;
  int cl = lane & 15;
  int rsel = (lane >> 4) * 8;

  Frag aggA[4], xA[4];
#pragma unroll
  for (int kb = 0; kb < 4; ++kb) {
    aggA[kb] = load_afrag_f32(agg, r0, kb, n_nodes);
    xA[kb]   = load_afrag_f32(x,   r0, kb, n_nodes);
  }

  // x in C/D accumulator layout, cached in VGPRs across both phases
  float xs[64];

  // ---- phase 1: rl = relu(agg@W2^T + x@U2^T + b2), store rl*x to LDS ----
#pragma unroll
  for (int nt = 0; nt < 8; ++nt) {
    f32x8 acc = {};
#pragma unroll
    for (int kb = 0; kb < 4; ++kb) {
      acc = wmma_bf16(aggA[kb], load_bfrag(w2, nt * 16, kb), acc);
      acc = wmma_bf16(xA[kb],   load_bfrag(u2, nt * 16, kb), acc);
    }
    int c = nt * 16 + cl;
    float bias = W2b[c] + U2b[c];
#pragma unroll
    for (int g = 0; g < 8; ++g) {
      int row = r0 + g + rsel;
      int rr = row < n_nodes ? row : n_nodes - 1;
      float xv = x[(size_t)rr * DFEAT + c];
      xs[nt * 8 + g] = xv;
      float rlv = fmaxf(acc[g] + bias, 0.0f);
      rlx[(g + rsel) * DFEAT + c] = rlv * xv;
    }
  }
  // LDS ops from the same wave are in-order (DScnt): loads below see stores.
  Frag rA[4];
#pragma unroll
  for (int kb = 0; kb < 4; ++kb) rA[kb] = load_afrag_lds(rlx, kb);

  // ---- phase 2: ul (W1/U1), fl (W3/U3 on rl*x), blend, BN1 stats --------
#pragma unroll
  for (int nt = 0; nt < 8; ++nt) {
    f32x8 a1 = {};
    f32x8 a3 = {};
#pragma unroll
    for (int kb = 0; kb < 4; ++kb) {
      a1 = wmma_bf16(aggA[kb], load_bfrag(w1, nt * 16, kb), a1);
      a1 = wmma_bf16(xA[kb],   load_bfrag(u1, nt * 16, kb), a1);
      a3 = wmma_bf16(aggA[kb], load_bfrag(w3, nt * 16, kb), a3);
      a3 = wmma_bf16(rA[kb],   load_bfrag(u3, nt * 16, kb), a3);
    }
    int c = nt * 16 + cl;
    float b1 = W1b[c] + U1b[c];
    float b3 = W3b[c] + U3b[c];
    float s = 0.0f, s2 = 0.0f;
#pragma unroll
    for (int g = 0; g < 8; ++g) {
      int row = r0 + g + rsel;
      if (row < n_nodes) {
        float xv = xs[nt * 8 + g];
        float ul = fmaxf(a1[g] + b1, 0.0f);
        float fl = tanhf(a3[g] + b3);
        float o = ul * fl + (1.0f - ul) * xv;
        out[(size_t)row * DFEAT + c] = o;
        s += o;
        s2 += o * o;
      }
    }
    // lanes l and l^16 hold the two 8-row halves of the same column
    s  += __shfl_xor(s, 16, 32);
    s2 += __shfl_xor(s2, 16, 32);
    if (lane < 16) {
      unsafeAtomicAdd(&bn_sum[c], s);
      unsafeAtomicAdd(&bn_sq[c], s2);
    }
  }
}

// ======================= K4/K6: BN coefficient finalize ===================
__global__ __launch_bounds__(128) void bncoef_kernel(
    const float* __restrict__ sum, const float* __restrict__ sq,
    const float* __restrict__ g, const float* __restrict__ b,
    float* __restrict__ a, float* __restrict__ c, float inv_n) {
  int i = threadIdx.x;
  if (i < DFEAT) {
    float m = sum[i] * inv_n;
    float v = sq[i] * inv_n - m * m;
    float s = g[i] * rsqrtf(v + 1e-5f);
    a[i] = s;
    c[i] = b[i] - m * s;
  }
}

// ======================= K5/K7: MLP layer (fused BN+ReLU input) ===========
// out[r,n] = relu(bn(in))[r,:] @ W^T + bias; optionally accumulate stats.
__global__ __launch_bounds__(32) void mlp_kernel(
    const float* __restrict__ in, const float* __restrict__ bn_a,
    const float* __restrict__ bn_c, const uint32_t* __restrict__ wq,
    const float* __restrict__ bias, float* __restrict__ outbuf,
    float* __restrict__ st_sum, float* __restrict__ st_sq, int n_nodes,
    int do_stats) {
  int r0 = blockIdx.x * 16;
  if (r0 >= n_nodes) return;
  int lane = threadIdx.x & 31;
  int cl = lane & 15;
  int rsel = (lane >> 4) * 8;

  Frag zA[4];
#pragma unroll
  for (int kb = 0; kb < 4; ++kb)
    zA[kb] = load_afrag_bnrelu(in, r0, kb, n_nodes, bn_a, bn_c);

#pragma unroll
  for (int nt = 0; nt < 8; ++nt) {
    f32x8 acc = {};
#pragma unroll
    for (int kb = 0; kb < 4; ++kb)
      acc = wmma_bf16(zA[kb], load_bfrag(wq, nt * 16, kb), acc);
    int c = nt * 16 + cl;
    float bv = bias[c];
    float s = 0.0f, s2 = 0.0f;
#pragma unroll
    for (int g = 0; g < 8; ++g) {
      int row = r0 + g + rsel;
      if (row < n_nodes) {
        float h = acc[g] + bv;
        outbuf[(size_t)row * DFEAT + c] = h;
        s += h;
        s2 += h * h;
      }
    }
    if (do_stats) {
      s  += __shfl_xor(s, 16, 32);
      s2 += __shfl_xor(s2, 16, 32);
      if (lane < 16) {
        unsafeAtomicAdd(&st_sum[c], s);
        unsafeAtomicAdd(&st_sq[c], s2);
      }
    }
  }
}

// ======================= host-side orchestration ==========================
extern "C" void kernel_launch(void* const* d_in, const int* in_sizes, int n_in,
                              void* d_out, int out_size, void* d_ws,
                              size_t ws_size, hipStream_t stream) {
  const float* x     = (const float*)d_in[0];
  const int*   erow  = (const int*)d_in[1];
  const int*   ecol  = (const int*)d_in[2];
  const float* eval_ = (const float*)d_in[3];
  const float* W1w = (const float*)d_in[4],  *W1b = (const float*)d_in[5];
  const float* U1w = (const float*)d_in[6],  *U1b = (const float*)d_in[7];
  const float* W2w = (const float*)d_in[8],  *W2b = (const float*)d_in[9];
  const float* U2w = (const float*)d_in[10], *U2b = (const float*)d_in[11];
  const float* W3w = (const float*)d_in[12], *W3b = (const float*)d_in[13];
  const float* U3w = (const float*)d_in[14], *U3b = (const float*)d_in[15];
  const float* bn_g = (const float*)d_in[16], *bn_b = (const float*)d_in[17];
  const float* m_w0 = (const float*)d_in[18], *m_b0 = (const float*)d_in[19];
  const float* m_bng = (const float*)d_in[20], *m_bnb = (const float*)d_in[21];
  const float* m_w1 = (const float*)d_in[22], *m_b1 = (const float*)d_in[23];

  const int n_nodes = in_sizes[0] / DFEAT;
  const int n_edges = in_sizes[1];

  // workspace layout
  char* ws = (char*)d_ws;
  float* agg = (float*)ws;  // [N,128] fp32; reused as `h` after gate kernel
  size_t off = (size_t)n_nodes * DFEAT * sizeof(float);
  uint32_t* wbf = (uint32_t*)(ws + off);  // 8 weight mats, bf16-pair packed
  const int WPAIRS = DFEAT * DFEAT / 2;   // 8192 u32 per matrix
  off += (size_t)8 * WPAIRS * sizeof(uint32_t);
  float* stats = (float*)(ws + off);  // bn1_sum, bn1_sq, bn2_sum, bn2_sq
  off += 4 * DFEAT * sizeof(float);
  float* coef = (float*)(ws + off);   // a1, c1, a2, c2

  uint32_t* wb[8];
  const float* wsrc[8] = {W1w, U1w, W2w, U2w, W3w, U3w, m_w0, m_w1};
  for (int i = 0; i < 8; ++i) wb[i] = wbf + (size_t)i * WPAIRS;

  // K0: convert weights to bf16 (L2-resident, 256 KB total)
  for (int i = 0; i < 8; ++i)
    cvt_bf16_kernel<<<(WPAIRS + 255) / 256, 256, 0, stream>>>(wsrc[i], wb[i],
                                                              WPAIRS);

  // K1: zero agg + BN stat accumulators
  int nz = n_nodes * DFEAT;
  zero_kernel<<<(nz + 255) / 256, 256, 0, stream>>>(agg, nz);
  zero_kernel<<<2, 256, 0, stream>>>(stats, 4 * DFEAT);

  // K2: SpMM scatter
  long sp_threads = (long)n_edges * 32;
  spmm_kernel<<<(int)((sp_threads + 255) / 256), 256, 0, stream>>>(
      x, erow, ecol, eval_, agg, n_edges);

  // K3: fused gate (6 WMMA GEMMs) -> d_out, BN1 stats
  int nwaves = (n_nodes + 15) / 16;
  gate_kernel<<<nwaves, 32, 0, stream>>>(
      x, agg, wb[0], wb[1], wb[2], wb[3], wb[4], wb[5], W1b, U1b, W2b, U2b,
      W3b, U3b, (float*)d_out, stats, stats + DFEAT, n_nodes);

  float inv_n = 1.0f / (float)n_nodes;
  // K4: BN1 coefficients
  bncoef_kernel<<<1, 128, 0, stream>>>(stats, stats + DFEAT, bn_g, bn_b, coef,
                                       coef + DFEAT, inv_n);

  // K5: h = relu(bn1(out)) @ m_w0^T + m_b0 (h overwrites agg), BN2 stats
  mlp_kernel<<<nwaves, 32, 0, stream>>>(
      (const float*)d_out, coef, coef + DFEAT, wb[6], m_b0, agg,
      stats + 2 * DFEAT, stats + 3 * DFEAT, n_nodes, 1);

  // K6: BN2 coefficients
  bncoef_kernel<<<1, 128, 0, stream>>>(stats + 2 * DFEAT, stats + 3 * DFEAT,
                                       m_bng, m_bnb, coef + 2 * DFEAT,
                                       coef + 3 * DFEAT, inv_n);

  // K7: d_out = relu(bn2(h)) @ m_w1^T + m_b1
  mlp_kernel<<<nwaves, 32, 0, stream>>>(
      agg, coef + 2 * DFEAT, coef + 3 * DFEAT, wb[7], m_b1, (float*)d_out,
      nullptr, nullptr, n_nodes, 0);
}
